// MyGPT2Attention_8839042695366
// MI455X (gfx1250) — matrix operand order repaired and checked by reading, not compiled, not run on hardware
//
#include <hip/hip_runtime.h>
#include <hip/hip_bf16.h>
#include <math.h>

// ---------------------------------------------------------------------------
// GPT-2 attention block on gfx1250 (MI455X), bf16 WMMA pipeline.
//   B=2, S=2048, D=1024, H=16, HD=64
// Round 2: 32x64 wave tiles in the GEMMs (1.5 b128 loads per WMMA instead of
// 2.5), global_prefetch of next K-step / next key chunk.
// ---------------------------------------------------------------------------

typedef __bf16 bf16_t;
typedef __attribute__((ext_vector_type(16))) __bf16 v16bf;
typedef __attribute__((ext_vector_type(8)))  __bf16 v8bf;
typedef __attribute__((ext_vector_type(8)))  float  v8f;

#define BB   2
#define SS   2048
#define DD   1024
#define HH   16
#define HD   64
#define MROWS (BB * SS)      // 4096
#define N_QKV (3 * DD)       // 3072

__device__ __forceinline__ v8f wmma_bf16(v16bf a, v16bf b, v8f c) {
  return __builtin_amdgcn_wmma_f32_16x16x32_bf16(
      /*neg_a=*/false, a, /*neg_b=*/false, b,
      /*c_mod=*/(short)0, c, /*reuse_a=*/false, /*reuse_b=*/false);
}

// A-fragment (16x32 bf16, row-major source): lane m = lane&15, half = lane>>4.
// Elements 0..15 <- K = half*16 + j (contiguous).
__device__ __forceinline__ v16bf load_a_frag(const bf16_t* row, int half) {
  v16bf a = *(const v16bf*)(row + half * 16);
  return a;
}

__device__ __forceinline__ float rowmax16(float v) {
#pragma unroll
  for (int m = 1; m < 16; m <<= 1) v = fmaxf(v, __shfl_xor(v, m, 32));
  return v;
}
__device__ __forceinline__ float rowsum16(float v) {
#pragma unroll
  for (int m = 1; m < 16; m <<= 1) v += __shfl_xor(v, m, 32);
  return v;
}

// ---------------------------------------------------------------------------
// Kernel 1: fp32 -> bf16 elementwise
// ---------------------------------------------------------------------------
__global__ void k_f32_to_bf16(const float* __restrict__ in, bf16_t* __restrict__ out, int n) {
  int i = blockIdx.x * blockDim.x + threadIdx.x;
  if (i < n) out[i] = (bf16_t)in[i];
}

// Kernel 2: transpose + convert: in [rows][cols] f32 -> out [cols][rows] bf16
__global__ void k_transpose_bf16(const float* __restrict__ in, bf16_t* __restrict__ out,
                                 int rows, int cols) {
  int i = blockIdx.x * blockDim.x + threadIdx.x;
  if (i >= rows * cols) return;
  int c = i / rows, r = i - c * rows;
  out[i] = (bf16_t)in[(size_t)r * cols + c];   // out[c*rows + r]
}

// ---------------------------------------------------------------------------
// Kernel 3: QKV GEMM.  X[4096,1024]bf16 x WaT[3072,1024]bf16 (+bias f32).
// Each wave: 32x64 output tile (2 M-subtiles x 4 N-subtiles, B-frags reused).
// Epilogue scatters into:
//   Qs [B*H][S][HD] (scaled by 1/sqrt(HD)), Ks [B*H][S][HD], Vt [B*H][HD][S]
// ---------------------------------------------------------------------------
__global__ void k_qkv_gemm(const bf16_t* __restrict__ X, const bf16_t* __restrict__ WaT,
                           const float* __restrict__ bias,
                           bf16_t* __restrict__ Qs, bf16_t* __restrict__ Ks,
                           bf16_t* __restrict__ Vt) {
  const int lane = threadIdx.x & 31;
  const int wid  = (blockIdx.x * blockDim.x + threadIdx.x) >> 5;
  const int NT   = N_QKV / 64;           // 48 N-tiles
  const int mt = wid / NT, nt = wid - mt * NT;
  const int row0 = mt * 32, col0 = nt * 64;
  const int half = lane >> 4, lan = lane & 15;

  v8f acc[2][4] = {};
  const bf16_t* arow0 = X + (size_t)(row0 + lan) * DD;
  const bf16_t* arow1 = X + (size_t)(row0 + 16 + lan) * DD;
  for (int k0 = 0; k0 < DD; k0 += 32) {
    // prefetch next K-step of the two A rows (global_prefetch_b8)
    if (k0 + 32 < DD) {
      __builtin_prefetch(arow0 + k0 + 32, 0, 3);
      __builtin_prefetch(arow1 + k0 + 32, 0, 3);
    }
    v16bf a0 = load_a_frag(arow0 + k0, half);
    v16bf a1 = load_a_frag(arow1 + k0, half);
#pragma unroll
    for (int t = 0; t < 4; ++t) {
      // B-frag: lane column n = lan, K = 16*half + j, contiguous in WaT row.
      v16bf b = *(const v16bf*)(WaT + (size_t)(col0 + t * 16 + lan) * DD + k0 + half * 16);
      acc[0][t] = wmma_bf16(a0, b, acc[0][t]);
      acc[1][t] = wmma_bf16(a1, b, acc[1][t]);
    }
  }
#pragma unroll
  for (int mi = 0; mi < 2; ++mi) {
#pragma unroll
    for (int t = 0; t < 4; ++t) {
#pragma unroll
      for (int r = 0; r < 8; ++r) {
        int m = row0 + mi * 16 + r + 8 * half;  // C layout: VGPR r -> M = r + 8*half
        int c = col0 + t * 16 + lan;            // N = lane&15
        float v = acc[mi][t][r] + bias[c];
        int b = m >> 11, s = m & (SS - 1);
        int which = c >> 10, cc = c & (DD - 1);
        int h = cc >> 6, d = cc & (HD - 1);
        int bh = b * HH + h;
        if (which == 0)
          Qs[((size_t)bh * SS + s) * HD + d] = (bf16_t)(v * 0.125f);  // fold 1/sqrt(64)
        else if (which == 1)
          Ks[((size_t)bh * SS + s) * HD + d] = (bf16_t)v;
        else
          Vt[((size_t)bh * HD + d) * SS + s] = (bf16_t)v;             // transposed V
      }
    }
  }
}

// ---------------------------------------------------------------------------
// Kernel 4: causal flash attention.  One wave = one (b,h, 16-query tile).
// Keys processed in 32-wide chunks; online softmax; P restaged via LDS to
// convert C-fragment layout -> A-fragment layout for the AV WMMAs.
// ---------------------------------------------------------------------------
__global__ void k_attn(const bf16_t* __restrict__ Qs, const bf16_t* __restrict__ Ks,
                       const bf16_t* __restrict__ Vt, bf16_t* __restrict__ Obf) {
  __shared__ __align__(32) bf16_t psm[4][16 * 32];   // per-wave 16x32 P tile
  const int lane = threadIdx.x & 31;
  const int widx = threadIdx.x >> 5;
  const int wid  = blockIdx.x * (blockDim.x >> 5) + widx;  // 0..4095
  const int qt = wid & (SS / 16 - 1);                       // 0..127
  const int bh = wid >> 7;                                  // 0..31
  const int q0 = qt * 16;
  const int half = lane >> 4, lan = lane & 15;

  const bf16_t* Qb = Qs + (size_t)bh * SS * HD;
  const bf16_t* Kb = Ks + (size_t)bh * SS * HD;
  const bf16_t* Vb = Vt + (size_t)bh * HD * SS;

  // Q A-fragments for d = 0..31 and 32..63 (reused over all key chunks)
  v16bf aq0 = load_a_frag(Qb + (size_t)(q0 + lan) * HD + 0,  half);
  v16bf aq1 = load_a_frag(Qb + (size_t)(q0 + lan) * HD + 32, half);

  v8f o[4] = {};
  float rmax[8], rsum[8];
#pragma unroll
  for (int r = 0; r < 8; ++r) { rmax[r] = -__builtin_inff(); rsum[r] = 0.f; }

  bf16_t* pw = &psm[widx][0];
  const int kend = q0 + 16;                  // causal: keys < q0+16
  for (int kc = 0; kc < kend; kc += 32) {
    // prefetch next chunk's K rows while this chunk computes
    if (kc + 32 < kend) {
      __builtin_prefetch(Kb + (size_t)(kc + 32 + lane) * HD, 0, 3);
    }
    // ---- scores: two 16x16 tiles (keys kc..kc+15 and kc+16..kc+31) ----
    v8f s0 = {}, s1 = {};
    {
      v16bf b00 = *(const v16bf*)(Kb + (size_t)(kc + lan) * HD + 0  + half * 16);
      v16bf b01 = *(const v16bf*)(Kb + (size_t)(kc + lan) * HD + 32 + half * 16);
      s0 = wmma_bf16(aq0, b00, s0);
      s0 = wmma_bf16(aq1, b01, s0);
      v16bf b10 = *(const v16bf*)(Kb + (size_t)(kc + 16 + lan) * HD + 0  + half * 16);
      v16bf b11 = *(const v16bf*)(Kb + (size_t)(kc + 16 + lan) * HD + 32 + half * 16);
      s1 = wmma_bf16(aq0, b10, s1);
      s1 = wmma_bf16(aq1, b11, s1);
    }
    // ---- causal mask + online softmax (row r+8*half lives in reg r) ----
    float p0[8], p1[8], sc[8];
#pragma unroll
    for (int r = 0; r < 8; ++r) {
      int q = q0 + r + 8 * half;
      float x0 = (kc + lan      <= q) ? s0[r] : -__builtin_inff();
      float x1 = (kc + 16 + lan <= q) ? s1[r] : -__builtin_inff();
      float cm = rowmax16(fmaxf(x0, x1));
      float nm = fmaxf(rmax[r], cm);
      p0[r] = __expf(x0 - nm);
      p1[r] = __expf(x1 - nm);
      float cs = rowsum16(p0[r] + p1[r]);
      sc[r]   = __expf(rmax[r] - nm);
      rsum[r] = rsum[r] * sc[r] + cs;
      rmax[r] = nm;
    }
#pragma unroll
    for (int t = 0; t < 4; ++t)
#pragma unroll
      for (int r = 0; r < 8; ++r) o[t][r] *= sc[r];

    // ---- stage P (C layout) into LDS, reload as A-fragment ----
#pragma unroll
    for (int r = 0; r < 8; ++r) {
      int row = r + 8 * half;
      pw[row * 32 + lan]      = (bf16_t)p0[r];
      pw[row * 32 + 16 + lan] = (bf16_t)p1[r];
    }
    __asm__ volatile("s_wait_dscnt 0" ::: "memory");   // wave-local RAW on LDS
    v16bf pa = load_a_frag(pw + lan * 32, half);

    // ---- O += P x V : 4 d-tiles of 16, V read from transposed layout ----
#pragma unroll
    for (int dt = 0; dt < 4; ++dt) {
      v16bf bv = *(const v16bf*)(Vb + (size_t)(dt * 16 + lan) * SS + kc + half * 16);
      o[dt] = wmma_bf16(pa, bv, o[dt]);
    }
  }

  // ---- finalize: O /= rowsum, merge heads into Obf[B*S][D] ----
  const int b = bh >> 4, h = bh & (HH - 1);
#pragma unroll
  for (int dt = 0; dt < 4; ++dt) {
#pragma unroll
    for (int r = 0; r < 8; ++r) {
      float v = o[dt][r] / rsum[r];
      int grow = b * SS + q0 + r + 8 * half;
      int gcol = h * HD + dt * 16 + lan;
      Obf[(size_t)grow * DD + gcol] = (bf16_t)v;
    }
  }
}

// ---------------------------------------------------------------------------
// Kernel 5: output projection.  O[4096,1024]bf16 x WpT[1024,1024]bf16 + bias
// -> fp32 d_out.  Same 32x64 wave tile as QKV GEMM.
// ---------------------------------------------------------------------------
__global__ void k_proj_gemm(const bf16_t* __restrict__ O, const bf16_t* __restrict__ WpT,
                            const float* __restrict__ bias, float* __restrict__ out) {
  const int lane = threadIdx.x & 31;
  const int wid  = (blockIdx.x * blockDim.x + threadIdx.x) >> 5;
  const int NT = DD / 64;                 // 16 N-tiles
  const int mt = wid / NT, nt = wid - mt * NT;
  const int row0 = mt * 32, col0 = nt * 64;
  const int half = lane >> 4, lan = lane & 15;

  v8f acc[2][4] = {};
  const bf16_t* arow0 = O + (size_t)(row0 + lan) * DD;
  const bf16_t* arow1 = O + (size_t)(row0 + 16 + lan) * DD;
  for (int k0 = 0; k0 < DD; k0 += 32) {
    if (k0 + 32 < DD) {
      __builtin_prefetch(arow0 + k0 + 32, 0, 3);
      __builtin_prefetch(arow1 + k0 + 32, 0, 3);
    }
    v16bf a0 = load_a_frag(arow0 + k0, half);
    v16bf a1 = load_a_frag(arow1 + k0, half);
#pragma unroll
    for (int t = 0; t < 4; ++t) {
      v16bf b = *(const v16bf*)(WpT + (size_t)(col0 + t * 16 + lan) * DD + k0 + half * 16);
      acc[0][t] = wmma_bf16(a0, b, acc[0][t]);
      acc[1][t] = wmma_bf16(a1, b, acc[1][t]);
    }
  }
#pragma unroll
  for (int mi = 0; mi < 2; ++mi) {
#pragma unroll
    for (int t = 0; t < 4; ++t) {
#pragma unroll
      for (int r = 0; r < 8; ++r) {
        int m = row0 + mi * 16 + r + 8 * half;
        int c = col0 + t * 16 + lan;
        out[(size_t)m * DD + c] = acc[mi][t][r] + bias[c];
      }
    }
  }
}

// ---------------------------------------------------------------------------
// Host launcher
// ---------------------------------------------------------------------------
extern "C" void kernel_launch(void* const* d_in, const int* in_sizes, int n_in,
                              void* d_out, int out_size, void* d_ws, size_t ws_size,
                              hipStream_t stream) {
  (void)in_sizes; (void)n_in; (void)out_size; (void)ws_size;
  const float* hs     = (const float*)d_in[0];   // [B,S,D]
  const float* w_attn = (const float*)d_in[1];   // [D,3D]
  const float* b_attn = (const float*)d_in[2];   // [3D]
  const float* w_proj = (const float*)d_in[3];   // [D,D]
  const float* b_proj = (const float*)d_in[4];   // [D]
  float* out = (float*)d_out;

  char* ws = (char*)d_ws;
  size_t off = 0;
  bf16_t* Xbf = (bf16_t*)(ws + off); off += (size_t)MROWS * DD * 2;        //  8 MB
  bf16_t* WaT = (bf16_t*)(ws + off); off += (size_t)N_QKV * DD * 2;        //  6 MB
  bf16_t* WpT = (bf16_t*)(ws + off); off += (size_t)DD * DD * 2;           //  2 MB
  bf16_t* Qsc = (bf16_t*)(ws + off); off += (size_t)BB * HH * SS * HD * 2; //  8 MB
  bf16_t* Ksc = (bf16_t*)(ws + off); off += (size_t)BB * HH * SS * HD * 2; //  8 MB
  bf16_t* Vtr = (bf16_t*)(ws + off); off += (size_t)BB * HH * HD * SS * 2; //  8 MB
  bf16_t* Obf = (bf16_t*)(ws + off); off += (size_t)MROWS * DD * 2;        //  8 MB

  // 1) hidden states -> bf16
  {
    int n = MROWS * DD;
    k_f32_to_bf16<<<(n + 255) / 256, 256, 0, stream>>>(hs, Xbf, n);
  }
  // 2) weight transposes -> bf16 (B-fragments become contiguous loads)
  {
    int n = DD * N_QKV;
    k_transpose_bf16<<<(n + 255) / 256, 256, 0, stream>>>(w_attn, WaT, DD, N_QKV);
  }
  {
    int n = DD * DD;
    k_transpose_bf16<<<(n + 255) / 256, 256, 0, stream>>>(w_proj, WpT, DD, DD);
  }
  // 3) QKV GEMM: 128 x 48 wave tiles (32x64 each), 8 waves per 256-thread block
  {
    int waves = (MROWS / 32) * (N_QKV / 64);   // 6144
    k_qkv_gemm<<<waves / 8, 256, 0, stream>>>(Xbf, WaT, b_attn, Qsc, Ksc, Vtr);
  }
  // 4) flash attention: B*H*(S/16) = 4096 waves, 4 waves per 128-thread block
  {
    int waves = BB * HH * (SS / 16);
    k_attn<<<waves / 4, 128, 0, stream>>>(Qsc, Ksc, Vtr, Obf);
  }
  // 5) output projection: 128 x 16 wave tiles (32x64 each)
  {
    int waves = (MROWS / 32) * (DD / 64);      // 2048
    k_proj_gemm<<<waves / 8, 256, 0, stream>>>(Obf, WpT, b_proj, out);
  }
}
